// SpectralEnvelopeShift_20220706029772
// MI455X (gfx1250) — matrix operand order repaired and checked by reading, not compiled, not run on hardware
//
#include <hip/hip_runtime.h>
#include <hip/hip_bf16.h>
#include <math.h>

// ---------------------------------------------------------------------------
// SpectralEnvelopeShift on MI455X (gfx1250):
//   STFT / iSTFT expressed as dense f16 GEMMs (WMMA 16x16x32, f32 accum).
//   N_FFT=512, HOP=128, PAD=256, rectangular window, reflect padding.
// ---------------------------------------------------------------------------

typedef __attribute__((ext_vector_type(16))) _Float16 v16h;
typedef __attribute__((ext_vector_type(8)))  _Float16 v8h;
typedef __attribute__((ext_vector_type(8)))  float    v8f;

#define NFFT   512
#define HOP    128
#define PADW   256
#define KT_CNT 16          // 512 / 32
#define NT_CNT 32          // 512 / 16

static __device__ __forceinline__ v16h cat8(v8h a, v8h b) {
    return __builtin_shufflevector(a, b, 0,1,2,3,4,5,6,7,8,9,10,11,12,13,14,15);
}

// ---------------------------------------------------------------------------
// Basis builder: writes both GEMM B-operands in WMMA B-fragment order.
// Fragment layout (dense 32x16 B, wave32, per ISA B layout):
//   lane L, half h(0..15):  K_local = 8*(L/16) + (h<8 ? h : 8+h),  N_local = L%16
// Flat index: (((nt*16 + kt)*32 + lane)*16 + h)
// basis1 (forward rFFT):  K = time n, N = packed col (257 Re ++ 255 Im)
// basis2 (inverse rFFT):  K = packed col, N = time n, scales folded in.
// ---------------------------------------------------------------------------
__global__ void build_basis(_Float16* __restrict__ b1, _Float16* __restrict__ b2) {
    int idx  = blockIdx.x * blockDim.x + threadIdx.x;   // < 512*512
    int h    = idx & 15;
    int lane = (idx >> 4) & 31;
    int kt   = (idx >> 9) & 15;
    int nt   = idx >> 13;
    int Kl   = (h < 8) ? ((lane >> 4) * 8 + h) : (16 + (lane >> 4) * 8 + h - 8);
    int Nl   = lane & 15;
    int K    = kt * 32 + Kl;                            // 0..511
    int N    = nt * 16 + Nl;                            // 0..511
    const float W = 6.28318530717958647692f / (float)NFFT;

    // forward: row K = time sample n, col N = packed bin
    {
        int n = K, col = N;
        float v;
        if (col < 257) {
            v = cosf((float)((col * n) & (NFFT - 1)) * W);
        } else {
            int kb = col - 256;                          // bins 1..255
            v = -sinf((float)((kb * n) & (NFFT - 1)) * W);
        }
        b1[idx] = (_Float16)v;
    }
    // inverse: row K = packed bin, col N = time sample n
    {
        int pk = K, n = N;
        float v;
        if (pk < 257) {
            float s = (pk == 0 || pk == 256) ? (1.0f / NFFT) : (2.0f / NFFT);
            v = cosf((float)((pk * n) & (NFFT - 1)) * W) * s;
        } else {
            int kb = pk - 256;
            v = -sinf((float)((kb * n) & (NFFT - 1)) * W) * (2.0f / NFFT);
        }
        b2[idx] = (_Float16)v;
    }
}

// ---------------------------------------------------------------------------
// Forward STFT GEMM.  One block = one M-tile (16 frames, 2560 samples staged
// into LDS as f16 with reflect padding).  4 waves, each owns 8 N-tiles.
// spec[frame*512 + col] (f32), col packed Re/Im.
// ---------------------------------------------------------------------------
__global__ __launch_bounds__(128) void stft_gemm(
        const float* __restrict__ w, int T, int n_frames,
        const _Float16* __restrict__ basis1, float* __restrict__ spec) {
    __shared__ _Float16 sx[2560];
    const int mtile = blockIdx.x;
    const int tid   = threadIdx.x;
    const int XP    = T + 2 * PADW;

    // Stage 16 overlapping frames (16*HOP + NFFT = 2560 samples), f32->f16,
    // inline reflect padding.
    for (int i = tid; i < 2560; i += 128) {
        int p = mtile * (16 * HOP) + i;                  // index into xp
        float v = 0.0f;
        if (p < XP) {
            int s = p - PADW;
            if (s < 0) s = -s;
            if (s >= T) s = 2 * T - 2 - s;
            v = w[s];
        }
        sx[i] = (_Float16)v;
    }
    __syncthreads();

    const int wave = tid >> 5;
    const int lane = tid & 31;
    const int M    = lane & 15;
    const int hgrp = (lane >> 4) ? 8 : 0;                // A K-half group
    const _Float16* arow = &sx[M * HOP];

    for (int nt = wave; nt < NT_CNT; nt += 4) {
        v8f c;
        #pragma unroll
        for (int r = 0; r < 8; ++r) c[r] = 0.0f;

        #pragma unroll
        for (int kt = 0; kt < KT_CNT; ++kt) {
            int kb = kt * 32 + hgrp;
            v8h a0 = *(const v8h*)(arow + kb);           // K = kb..kb+7
            v8h a1 = *(const v8h*)(arow + kb + 16);      // K = kb+16..kb+23
            v16h a = cat8(a0, a1);
            const _Float16* bp = basis1 + (((nt * 16 + kt) * 32 + lane) << 4);
            v16h b = cat8(*(const v8h*)bp, *(const v8h*)(bp + 8));
            c = __builtin_amdgcn_wmma_f32_16x16x32_f16(
                    false, a, false, b, (short)0, c, false, false);
        }
        // C layout: lanes 0-15: M=r, N=lane; lanes 16-31: M=8+r, N=lane-16
        int n    = lane & 15;
        int mofs = (lane >> 4) * 8;
        #pragma unroll
        for (int r = 0; r < 8; ++r) {
            int fr = mtile * 16 + mofs + r;              // may be pad frame: ok
            spec[fr * NFFT + nt * 16 + n] = c[r];
        }
    }
}

// ---------------------------------------------------------------------------
// Magnitude resample + recombination with original phase.
//   shifted = spec * (new_mag / mag)   (== new_mag * exp(i*angle(spec)))
// Writes f16 packed rows (A-operand of the inverse GEMM); pad frames -> 0.
// ---------------------------------------------------------------------------
__global__ void shift_kernel(const float* __restrict__ spec,
                             _Float16* __restrict__ spec2,
                             const float* __restrict__ shift_factor,
                             int n_frames, int mpad) {
    int idx = blockIdx.x * blockDim.x + threadIdx.x;
    if (idx >= mpad * NFFT) return;
    int f  = idx >> 9;
    int pk = idx & (NFFT - 1);
    if (f >= n_frames) { spec2[idx] = (_Float16)0.0f; return; }

    int kbin, is_re;
    if (pk < 257) { kbin = pk;       is_re = 1; }
    else          { kbin = pk - 256; is_re = 0; }

    const float* row = spec + f * NFFT;
    float re = row[kbin];
    float im = (kbin == 0 || kbin == 256) ? 0.0f : row[256 + kbin];
    float mag = sqrtf(re * re + im * im);

    float sf  = shift_factor[0];
    float src = (float)kbin / sf;
    int   lo  = (int)floorf(src);
    float fr  = src - (float)lo;
    int   hi  = lo + 1;
    float wlo = (lo < 257) ? (1.0f - fr) : 0.0f;
    float whi = (hi < 257) ? fr : 0.0f;
    int loc = lo < 256 ? lo : 256;
    int hic = hi < 256 ? hi : 256;

    float rl = row[loc];
    float il = (loc == 0 || loc == 256) ? 0.0f : row[256 + loc];
    float rh = row[hic];
    float ih = (hic == 0 || hic == 256) ? 0.0f : row[256 + hic];
    float nm = wlo * sqrtf(rl * rl + il * il) + whi * sqrtf(rh * rh + ih * ih);

    float outv;
    if (mag > 1e-20f) outv = (is_re ? re : im) * (nm / mag);
    else              outv = is_re ? nm : 0.0f;          // angle(0)=0
    spec2[idx] = (_Float16)outv;
}

// ---------------------------------------------------------------------------
// Inverse STFT GEMM.  One block = one M-tile; A rows (16x512 f16 = 16 KB)
// staged in LDS.  Writes time-domain frames (reuses the spec f32 buffer).
// ---------------------------------------------------------------------------
__global__ __launch_bounds__(128) void istft_gemm(
        const _Float16* __restrict__ spec2,
        const _Float16* __restrict__ basis2,
        float* __restrict__ frames) {
    __shared__ _Float16 sa[16 * NFFT];
    const int mtile = blockIdx.x;
    const int tid   = threadIdx.x;

    const _Float16* src = spec2 + (long)mtile * 16 * NFFT;
    for (int i = tid * 8; i < 16 * NFFT; i += 128 * 8)
        *(v8h*)&sa[i] = *(const v8h*)&src[i];
    __syncthreads();

    const int wave = tid >> 5;
    const int lane = tid & 31;
    const int M    = lane & 15;
    const int hgrp = (lane >> 4) ? 8 : 0;
    const _Float16* arow = &sa[M * NFFT];

    for (int nt = wave; nt < NT_CNT; nt += 4) {
        v8f c;
        #pragma unroll
        for (int r = 0; r < 8; ++r) c[r] = 0.0f;

        #pragma unroll
        for (int kt = 0; kt < KT_CNT; ++kt) {
            int kb = kt * 32 + hgrp;
            v16h a = cat8(*(const v8h*)(arow + kb), *(const v8h*)(arow + kb + 16));
            const _Float16* bp = basis2 + (((nt * 16 + kt) * 32 + lane) << 4);
            v16h b = cat8(*(const v8h*)bp, *(const v8h*)(bp + 8));
            c = __builtin_amdgcn_wmma_f32_16x16x32_f16(
                    false, a, false, b, (short)0, c, false, false);
        }
        int n    = lane & 15;
        int mofs = (lane >> 4) * 8;
        #pragma unroll
        for (int r = 0; r < 8; ++r) {
            int fr = mtile * 16 + mofs + r;
            frames[fr * NFFT + nt * 16 + n] = c[r];
        }
    }
}

// ---------------------------------------------------------------------------
// Deterministic overlap-add (<=4 overlapping frames for HOP=128, NFFT=512)
// + rectangular-window envelope division + center crop.
// ---------------------------------------------------------------------------
__global__ void finalize_kernel(const float* __restrict__ frames,
                                float* __restrict__ out,
                                int T, int n_frames) {
    int i = blockIdx.x * blockDim.x + threadIdx.x;
    if (i >= T) return;
    int t = i + PADW;
    int fmax = t >> 7; if (fmax > n_frames - 1) fmax = n_frames - 1;
    int fmin = (t >= 384) ? ((t - 384) >> 7) : 0;   // ceil((t-511)/128), clamped
    float acc = 0.0f;
    for (int f = fmin; f <= fmax; ++f)
        acc += frames[f * NFFT + (t - f * HOP)];
    float den = (float)(fmax - fmin + 1);
    out[i] = acc / fmaxf(den, 1e-11f);
}

// ---------------------------------------------------------------------------
extern "C" void kernel_launch(void* const* d_in, const int* in_sizes, int n_in,
                              void* d_out, int out_size, void* d_ws, size_t ws_size,
                              hipStream_t stream) {
    const float* w  = (const float*)d_in[0];
    const float* sf = (const float*)d_in[1];
    float* out = (float*)d_out;

    const int T        = in_sizes[0];
    const int XP       = T + 2 * PADW;
    const int n_frames = 1 + (XP - NFFT) / HOP;
    const int mtiles   = (n_frames + 15) / 16;
    const int mpad     = mtiles * 16;

    // Workspace layout (all 16B-aligned by construction):
    char* ws = (char*)d_ws;
    _Float16* basis1 = (_Float16*)ws;                        // 512*512 f16
    _Float16* basis2 = basis1 + 512 * 512;                   // 512*512 f16
    float*    spec   = (float*)(basis2 + 512 * 512);         // mpad*512 f32
    _Float16* spec2  = (_Float16*)(spec + (long)mpad * NFFT);// mpad*512 f16
    float*    frames = spec;                                 // reuse: spec dead after shift

    build_basis<<<(512 * 512) / 256, 256, 0, stream>>>(basis1, basis2);
    stft_gemm<<<mtiles, 128, 0, stream>>>(w, T, n_frames, basis1, spec);
    {
        long total = (long)mpad * NFFT;
        shift_kernel<<<(int)((total + 255) / 256), 256, 0, stream>>>(
            spec, spec2, sf, n_frames, mpad);
    }
    istft_gemm<<<mtiles, 128, 0, stream>>>(spec2, basis2, frames);
    finalize_kernel<<<(T + 255) / 256, 256, 0, stream>>>(frames, out, T, n_frames);
}
